// GATRegressor_6038724018187
// MI455X (gfx1250) — matrix operand order repaired and checked by reading, not compiled, not run on hardware
//
#include <hip/hip_runtime.h>
#include <hip/hip_bf16.h>

typedef __attribute__((ext_vector_type(16))) _Float16 v16h;
typedef __attribute__((ext_vector_type(8)))  _Float16 v8h;
typedef __attribute__((ext_vector_type(8)))  float    v8f;

#define HID 64
#define IN_F 16
#define NEG_SLOPE 0.2f

// ---------------------------------------------------------------------------
// Helpers
// ---------------------------------------------------------------------------
__device__ __forceinline__ void edge_sd(const int* __restrict__ ei, int e, int E,
                                        int& s, int& d) {
    if (e < E) { s = ei[e]; d = ei[E + e]; }
    else       { s = e - E; d = e - E; }          // appended self-loops
}

// float atomic max via ordered integer bit patterns
__device__ __forceinline__ void atomicMaxFloat(float* addr, float val) {
    if (val >= 0.0f) atomicMax((int*)addr, __float_as_int(val));
    else             atomicMin((unsigned int*)addr, __float_as_uint(val));
}

__device__ __forceinline__ float wave_reduce_add(float v) {
    #pragma unroll
    for (int off = 16; off > 0; off >>= 1) v += __shfl_down(v, off, 32);
    return v;
}

// ---------------------------------------------------------------------------
// f32 -> f16 conversion (with ReLU) for activations
// ---------------------------------------------------------------------------
__global__ void k_cvt_relu_f16(const float* __restrict__ in,
                               _Float16* __restrict__ out, int n) {
    int i = blockIdx.x * blockDim.x + threadIdx.x;
    if (i < n) out[i] = (_Float16)fmaxf(in[i], 0.0f);
}

// ---------------------------------------------------------------------------
// Weight f32[K,64] -> f16 in WMMA B-fragment order.
// Layout: frag[((kb*4 + nTile)*32 + lane)*16 + j] = W[kb*32 + bKoff(lane) + j + (j>=8 ? 8 : 0)]
//                                                    [nTile*16 + (lane&15)]
// so a lane's 16-half fragment is one contiguous 32-byte chunk.
// ---------------------------------------------------------------------------
__global__ void k_cvt_w_frag(const float* __restrict__ W,
                             _Float16* __restrict__ out, int K, int total) {
    int idx = blockIdx.x * blockDim.x + threadIdx.x;
    if (idx >= total) return;
    int kb    = idx >> 11;          // 2048 halves per 32-K block
    int rem   = idx & 2047;
    int nT    = rem >> 9;           // 512 halves per nTile
    int rem2  = rem & 511;
    int lane  = rem2 >> 4;
    int j     = rem2 & 15;
    int bKoff = (lane < 16) ? 0 : 8;
    int k     = kb * 32 + bKoff + j + ((j >= 8) ? 8 : 0);
    int col   = nT * 16 + (lane & 15);
    out[idx]  = (k < K) ? (_Float16)W[k * HID + col] : (_Float16)0.0f;
}

// ---------------------------------------------------------------------------
// WMMA GEMM: C[M,64] = A[M,K](f16) x W(f16, fragment order), f32 accumulate.
// One wave32 per 16x16 tile. K compile-time (16 or 64) => no predicated
// loads; A fragment = two b128 loads, B fragment = one 32B load.
// Requires M % 16 == 0 (holds: N_NODES = 100000 = 6250 * 16), so the whole
// 16x16 C tile is stored unconditionally (no predicated store ladder).
// Layouts per CDNA5 ISA 7.12.2.
// ---------------------------------------------------------------------------
template<int K>
__global__ void k_gemm_wmma(const _Float16* __restrict__ A,
                            const _Float16* __restrict__ Bfrag,
                            float* __restrict__ C, int M) {
    constexpr int KB = (K + 31) / 32;
    const int wave  = threadIdx.x >> 5;
    const int lane  = threadIdx.x & 31;
    const int wid   = blockIdx.x * (blockDim.x >> 5) + wave;
    const int mTile = wid >> 2;
    const int nTile = wid & 3;
    if (mTile * 16 >= M) return;

    const int halfSel = lane >> 4;                 // 0: lanes 0-15, 1: 16-31
    const int row     = mTile * 16 + (lane & 15);
    const int aKoff   = halfSel ? 8 : 0;
    const _Float16* __restrict__ arow = A + (size_t)row * K;

    v8f acc = {};
    #pragma unroll
    for (int kb = 0; kb < KB; ++kb) {
        const int k0 = kb * 32;
        v8h g1 = {}, g2 = {};
        if constexpr ((K % 32) == 0) {
            g1 = *(const v8h*)(arow + k0 + aKoff);        // K = k0+aKoff..+7
            g2 = *(const v8h*)(arow + k0 + aKoff + 16);   // K = +16..+23
        } else {                                          // K == 16 path
            g1 = *(const v8h*)(arow + aKoff);             // valid halves
            // g2 stays zero: K-range 16..31 is padding
        }
        v16h a;
        #pragma unroll
        for (int j = 0; j < 8; ++j) { a[j] = g1[j]; a[8 + j] = g2[j]; }

        v16h b = *(const v16h*)(Bfrag + ((((kb * 4 + nTile) << 5) + lane) << 4));

        acc = __builtin_amdgcn_wmma_f32_16x16x32_f16(
            /*neg_a=*/false, a, /*neg_b=*/false, b,
            /*c_mod=*/(short)0, acc, /*reuse_a=*/false, /*reuse_b=*/false);
    }

    // Unconditional 16x16 tile store (M % 16 == 0 guaranteed by launcher).
    float* __restrict__ crow =
        C + (size_t)(mTile * 16 + (halfSel ? 8 : 0)) * HID
          + nTile * 16 + (lane & 15);
    #pragma unroll
    for (int r = 0; r < 8; ++r) {
        crow[(size_t)r * HID] = acc[r];
    }
}

// ---------------------------------------------------------------------------
// Attention dot products: a_src[i] = h[i,:]·att_src ; a_dst[i] = h[i,:]·att_dst
// One wave per node.
// ---------------------------------------------------------------------------
__global__ void k_att_dots(const float* __restrict__ h,
                           const float* __restrict__ att_s,
                           const float* __restrict__ att_d,
                           float* __restrict__ a_src,
                           float* __restrict__ a_dst, int N) {
    int gid  = blockIdx.x * blockDim.x + threadIdx.x;
    int node = gid >> 5;
    int lane = gid & 31;
    if (node >= N) return;
    float h0 = h[(size_t)node * HID + lane];
    float h1 = h[(size_t)node * HID + lane + 32];
    float s = h0 * att_s[lane] + h1 * att_s[lane + 32];
    float d = h0 * att_d[lane] + h1 * att_d[lane + 32];
    s = wave_reduce_add(s);
    d = wave_reduce_add(d);
    if (lane == 0) { a_src[node] = s; a_dst[node] = d; }
}

// ---------------------------------------------------------------------------
// Init kernels
// ---------------------------------------------------------------------------
__global__ void k_init_seg(float* __restrict__ seg_m, float* __restrict__ seg_d,
                           int N) {
    int i = blockIdx.x * blockDim.x + threadIdx.x;
    if (i < N) { seg_m[i] = -INFINITY; seg_d[i] = 0.0f; }
}

__global__ void k_init_bias(float* __restrict__ out, const float* __restrict__ b,
                            int N) {
    int i = blockIdx.x * blockDim.x + threadIdx.x;
    if (i < N * HID) out[i] = b[i & (HID - 1)];
}

__global__ void k_init_pool(float* __restrict__ gmax, float* __restrict__ gsum,
                            float* __restrict__ cnt, int G) {
    int i = blockIdx.x * blockDim.x + threadIdx.x;
    if (i < G * HID) { gmax[i] = -INFINITY; gsum[i] = 0.0f; }
    if (i < G) cnt[i] = 0.0f;
}

// ---------------------------------------------------------------------------
// Edge pass 1: segment max of leaky_relu(a_src[s] + a_dst[d]) over dst
// ---------------------------------------------------------------------------
__global__ void k_edge_max(const int* __restrict__ ei, int E, int ET,
                           const float* __restrict__ a_src,
                           const float* __restrict__ a_dst,
                           float* __restrict__ seg_m) {
    int e = blockIdx.x * blockDim.x + threadIdx.x;
    if (e >= ET) return;
    int s, d; edge_sd(ei, e, E, s, d);
    float l = a_src[s] + a_dst[d];
    l = (l > 0.0f) ? l : NEG_SLOPE * l;
    atomicMaxFloat(&seg_m[d], l);
}

__global__ void k_fix_m(float* __restrict__ seg_m, int N) {
    int i = blockIdx.x * blockDim.x + threadIdx.x;
    if (i < N) { float m = seg_m[i]; if (!isfinite(m)) seg_m[i] = 0.0f; }
}

// ---------------------------------------------------------------------------
// Edge pass 2: e = exp(l - m[d]); cache per-edge e; segment-sum into seg_d
// ---------------------------------------------------------------------------
__global__ void k_edge_expsum(const int* __restrict__ ei, int E, int ET,
                              const float* __restrict__ a_src,
                              const float* __restrict__ a_dst,
                              const float* __restrict__ seg_m,
                              float* __restrict__ edge_e,
                              float* __restrict__ seg_d) {
    int e = blockIdx.x * blockDim.x + threadIdx.x;
    if (e >= ET) return;
    int s, d; edge_sd(ei, e, E, s, d);
    float l = a_src[s] + a_dst[d];
    l = (l > 0.0f) ? l : NEG_SLOPE * l;
    float ev = __expf(l - seg_m[d]);
    edge_e[e] = ev;
    atomicAdd(&seg_d[d], ev);
}

// ---------------------------------------------------------------------------
// Edge pass 3: out[d,:] += h[s,:] * alpha  (wave32 per edge; lane handles
// channels {lane, lane+32} -> coalesced 256B row gather + f32 atomics in L2)
// ---------------------------------------------------------------------------
__global__ void k_edge_scatter(const int* __restrict__ ei, int E, int ET,
                               const float* __restrict__ h,
                               const float* __restrict__ edge_e,
                               const float* __restrict__ seg_d,
                               float* __restrict__ out) {
    int gid  = blockIdx.x * blockDim.x + threadIdx.x;
    int e    = gid >> 5;
    int lane = gid & 31;
    if (e >= ET) return;
    int s, d; edge_sd(ei, e, E, s, d);
    float alpha = edge_e[e] / (seg_d[d] + 1e-16f);
    float v0 = h[(size_t)s * HID + lane]      * alpha;
    float v1 = h[(size_t)s * HID + lane + 32] * alpha;
    atomicAdd(&out[(size_t)d * HID + lane],      v0);
    atomicAdd(&out[(size_t)d * HID + lane + 32], v1);
}

// ---------------------------------------------------------------------------
// Pooling: per-graph max + sum of relu(h); wave per node
// ---------------------------------------------------------------------------
__global__ void k_pool(const float* __restrict__ h, const int* __restrict__ batch,
                       float* __restrict__ gmax, float* __restrict__ gsum,
                       float* __restrict__ cnt, int N) {
    int gid  = blockIdx.x * blockDim.x + threadIdx.x;
    int node = gid >> 5;
    int lane = gid & 31;
    if (node >= N) return;
    int g = batch[node];
    float v0 = fmaxf(h[(size_t)node * HID + lane],      0.0f);
    float v1 = fmaxf(h[(size_t)node * HID + lane + 32], 0.0f);
    atomicMaxFloat(&gmax[(size_t)g * HID + lane],      v0);
    atomicMaxFloat(&gmax[(size_t)g * HID + lane + 32], v1);
    atomicAdd(&gsum[(size_t)g * HID + lane],      v0);
    atomicAdd(&gsum[(size_t)g * HID + lane + 32], v1);
    if (lane == 0) atomicAdd(&cnt[g], 1.0f);
}

// ---------------------------------------------------------------------------
// Final FC: out[g] = [gmax|gmean]·fc_w + fc_b ; wave per graph
// ---------------------------------------------------------------------------
__global__ void k_fc(const float* __restrict__ gmax, const float* __restrict__ gsum,
                     const float* __restrict__ cnt, const float* __restrict__ fc_w,
                     const float* __restrict__ fc_b, float* __restrict__ out, int G) {
    int gid  = blockIdx.x * blockDim.x + threadIdx.x;
    int g    = gid >> 5;
    int lane = gid & 31;
    if (g >= G) return;
    float c   = cnt[g];
    float inv = 1.0f / fmaxf(c, 1.0f);
    float acc = 0.0f;
    #pragma unroll
    for (int j = lane; j < HID; j += 32) {
        float mx = (c > 0.0f) ? gmax[(size_t)g * HID + j] : 0.0f;
        float mn = gsum[(size_t)g * HID + j] * inv;
        acc += mx * fc_w[j] + mn * fc_w[HID + j];
    }
    acc = wave_reduce_add(acc);
    if (lane == 0) out[g] = acc + fc_b[0];
}

// ---------------------------------------------------------------------------
// Host orchestration
// ---------------------------------------------------------------------------
extern "C" void kernel_launch(void* const* d_in, const int* in_sizes, int n_in,
                              void* d_out, int out_size, void* d_ws, size_t ws_size,
                              hipStream_t stream) {
    const float* x        = (const float*)d_in[0];
    const int*   ei       = (const int*)  d_in[1];
    const int*   batch    = (const int*)  d_in[2];
    const float* W1       = (const float*)d_in[3];
    const float* att_s1   = (const float*)d_in[4];
    const float* att_d1   = (const float*)d_in[5];
    const float* b1       = (const float*)d_in[6];
    const float* W2       = (const float*)d_in[7];
    const float* att_s2   = (const float*)d_in[8];
    const float* att_d2   = (const float*)d_in[9];
    const float* b2       = (const float*)d_in[10];
    const float* fc_w     = (const float*)d_in[11];
    const float* fc_b     = (const float*)d_in[12];
    float*       out      = (float*)d_out;

    const int N  = in_sizes[0] / IN_F;       // 100000 (multiple of 16)
    const int E  = in_sizes[1] / 2;          // 1600000
    const int ET = E + N;                    // + self loops
    const int G  = out_size;                 // 512

    // workspace carve-out (256B aligned slices)
    char* ws = (char*)d_ws;
    auto carve = [&](size_t bytes) {
        void* p = (void*)ws;
        ws += (bytes + 255) & ~((size_t)255);
        return p;
    };
    float*    h_feat = (float*)   carve((size_t)N * HID * sizeof(float));
    float*    h_out  = (float*)   carve((size_t)N * HID * sizeof(float));
    float*    a_src  = (float*)   carve((size_t)N * sizeof(float));
    float*    a_dst  = (float*)   carve((size_t)N * sizeof(float));
    float*    seg_m  = (float*)   carve((size_t)N * sizeof(float));
    float*    seg_d  = (float*)   carve((size_t)N * sizeof(float));
    float*    edge_e = (float*)   carve((size_t)ET * sizeof(float));
    _Float16* h16    = (_Float16*)carve((size_t)N * HID * sizeof(_Float16));
    _Float16* wfrag  = (_Float16*)carve((size_t)2 * 2048 * sizeof(_Float16));
    float*    gmax   = (float*)   carve((size_t)G * HID * sizeof(float));
    float*    gsum   = (float*)   carve((size_t)G * HID * sizeof(float));
    float*    cnt    = (float*)   carve((size_t)G * sizeof(float));
    (void)ws_size; (void)n_in;

    const int B = 256;
    auto grid1 = [&](long long n) { return (int)((n + B - 1) / B); };
    const int numWaves  = ((N + 15) / 16) * (HID / 16);
    const int gemmGrid  = (numWaves + 7) / 8;

    // ---------------- Layer 1 ----------------
    k_cvt_relu_f16<<<grid1((long long)N * IN_F), B, 0, stream>>>(x, h16, N * IN_F);
    k_cvt_w_frag  <<<grid1(1 * 2048),            B, 0, stream>>>(W1, wfrag, IN_F,
                                                                 1 * 2048);
    k_gemm_wmma<IN_F><<<gemmGrid, B, 0, stream>>>(h16, wfrag, h_feat, N);

    k_att_dots <<<grid1((long long)N * 32), B, 0, stream>>>(h_feat, att_s1, att_d1,
                                                            a_src, a_dst, N);
    k_init_seg <<<grid1(N),                 B, 0, stream>>>(seg_m, seg_d, N);
    k_init_bias<<<grid1((long long)N * HID),B, 0, stream>>>(h_out, b1, N);
    k_edge_max <<<grid1(ET),                B, 0, stream>>>(ei, E, ET, a_src, a_dst, seg_m);
    k_fix_m    <<<grid1(N),                 B, 0, stream>>>(seg_m, N);
    k_edge_expsum<<<grid1(ET),              B, 0, stream>>>(ei, E, ET, a_src, a_dst,
                                                            seg_m, edge_e, seg_d);
    k_edge_scatter<<<grid1((long long)ET * 32), B, 0, stream>>>(ei, E, ET, h_feat,
                                                                edge_e, seg_d, h_out);

    // ---------------- Layer 2 ----------------
    k_cvt_relu_f16<<<grid1((long long)N * HID), B, 0, stream>>>(h_out, h16, N * HID);
    k_cvt_w_frag  <<<grid1(2 * 2048),           B, 0, stream>>>(W2, wfrag, HID,
                                                                2 * 2048);
    k_gemm_wmma<HID><<<gemmGrid, B, 0, stream>>>(h16, wfrag, h_feat, N);

    k_att_dots <<<grid1((long long)N * 32), B, 0, stream>>>(h_feat, att_s2, att_d2,
                                                            a_src, a_dst, N);
    k_init_seg <<<grid1(N),                 B, 0, stream>>>(seg_m, seg_d, N);
    k_init_bias<<<grid1((long long)N * HID),B, 0, stream>>>(h_out, b2, N);
    k_edge_max <<<grid1(ET),                B, 0, stream>>>(ei, E, ET, a_src, a_dst, seg_m);
    k_fix_m    <<<grid1(N),                 B, 0, stream>>>(seg_m, N);
    k_edge_expsum<<<grid1(ET),              B, 0, stream>>>(ei, E, ET, a_src, a_dst,
                                                            seg_m, edge_e, seg_d);
    k_edge_scatter<<<grid1((long long)ET * 32), B, 0, stream>>>(ei, E, ET, h_feat,
                                                                edge_e, seg_d, h_out);

    // ---------------- Pool + FC ----------------
    k_init_pool<<<grid1((long long)G * HID), B, 0, stream>>>(gmax, gsum, cnt, G);
    k_pool     <<<grid1((long long)N * 32),  B, 0, stream>>>(h_out, batch, gmax, gsum,
                                                             cnt, N);
    k_fc       <<<grid1((long long)G * 32),  B, 0, stream>>>(gmax, gsum, cnt, fc_w,
                                                             fc_b, out, G);
}
